// neuron_layer_bntt_v2_8469675507799
// MI455X (gfx1250) — matrix operand order repaired and checked by reading, not compiled, not run on hardware
//
#include <hip/hip_runtime.h>
#include <hip/hip_bf16.h>

// ---------------------------------------------------------------------------
// Problem constants (from reference): B=128, D_IN=2048, N=2048, T=50
// ---------------------------------------------------------------------------
#define BB 128
#define DIN 2048
#define NN 2048
#define TT 50
#define MROWS (TT * BB)          // 6400 rows of the fused GEMM
#define BN_EPS 1e-4f

typedef float v8f __attribute__((ext_vector_type(8)));
typedef __bf16 bf16_t;
typedef bf16_t v16bf __attribute__((ext_vector_type(16)));

union FragU {
    uint4 q[2];
    v16bf v;
};

__device__ __forceinline__ unsigned short f2bf(float f) {
    unsigned int u = __float_as_uint(f);
    unsigned int r = u + 0x7FFFu + ((u >> 16) & 1u);   // round-to-nearest-even
    return (unsigned short)(r >> 16);
}

// --- CDNA5 async global->LDS copy (ASYNCcnt path), 16 bytes per lane --------
__device__ __forceinline__ void async_copy_b128(unsigned lds_off, const void* gptr) {
    asm volatile("global_load_async_to_lds_b128 %0, %1, off"
                 :: "v"(lds_off), "v"((unsigned long long)gptr)
                 : "memory");
}
__device__ __forceinline__ void wait_async0() {
    asm volatile("s_wait_asynccnt 0x0" ::: "memory");
}
__device__ __forceinline__ void wait_dscnt0() {
    asm volatile("s_wait_dscnt 0x0" ::: "memory");
}

// ---------------------------------------------------------------------------
// Kernel 1: W [N][DIN] f32 -> bf16 (row-major [n][k]: already the layout the
// WMMA B-fragment wants -- 32 contiguous K per fragment row)
// ---------------------------------------------------------------------------
__global__ void snn_pack_w(const float* __restrict__ W,
                           unsigned short* __restrict__ Wbf, int count) {
    int i = blockIdx.x * 256 + threadIdx.x;
    if (i < count) Wbf[i] = f2bf(W[i]);
}

// ---------------------------------------------------------------------------
// Kernel 2: input_spikes [B][DIN][T] f32 -> Xbf [(t*B+b)][d] bf16
// LDS 32x32 tile transpose of the (d,t) plane per batch element.
// grid: (DIN/32, ceil(T/32)=2, B), block: (32, 8)
// ---------------------------------------------------------------------------
__global__ void snn_pack_x(const float* __restrict__ in,
                           unsigned short* __restrict__ Xbf) {
    __shared__ float tile[32][33];
    const int d0 = blockIdx.x * 32;
    const int t0 = blockIdx.y * 32;
    const int b  = blockIdx.z;
    const int tx = threadIdx.x, ty = threadIdx.y;
    #pragma unroll
    for (int j = 0; j < 32; j += 8) {
        int d = d0 + ty + j;
        int t = t0 + tx;
        float v = 0.0f;
        if (t < TT) v = in[((size_t)b * DIN + d) * TT + t];  // coalesced in t
        tile[ty + j][tx] = v;
    }
    __syncthreads();
    #pragma unroll
    for (int j = 0; j < 32; j += 8) {
        int t = t0 + ty + j;
        int d = d0 + tx;
        if (t < TT)                                           // coalesced in d
            Xbf[((size_t)t * BB + b) * DIN + d] = f2bf(tile[tx][ty + j]);
    }
}

// ---------------------------------------------------------------------------
// Kernel 3: fused GEMM  psp[r][n] = sum_k Xbf[r][k] * Wbf[n][k] + bias[n]
// Block tile 128(M) x 128(N) x 64(K), 256 threads = 8 waves in a 4x2 grid.
// Each wave: 32x64 output tile = 2x4 accumulators, 16 WMMAs per K-tile.
// Double-buffered LDS filled by global_load_async_to_lds_b128 (ASYNCcnt);
// rows padded to 144 B (16B-aligned) so fragments are paired ds_load_b128.
// ---------------------------------------------------------------------------
#define LDSROW 72                 // ushorts per row (144 B), holds 64 K-elems
#define TILE_USH (128 * LDSROW)   // 9216 ushorts per tile buffer

__global__ __launch_bounds__(256)
void snn_gemm_psp(const unsigned short* __restrict__ Xbf,
                  const unsigned short* __restrict__ Wbf,
                  const float* __restrict__ bias,
                  float* __restrict__ psp) {
    __shared__ __align__(16) unsigned short lA[2][TILE_USH];
    __shared__ __align__(16) unsigned short lB[2][TILE_USH];

    const int tid  = threadIdx.x;
    const int lane = tid & 31;
    const int wave = tid >> 5;
    const int wy   = wave >> 1;      // 0..3  (32 rows each)
    const int wx   = wave & 1;       // 0..1  (64 cols each)
    const int l15  = lane & 15;
    const int g    = lane >> 4;      // half-wave select

    const int n0 = blockIdx.x * 128;
    const int r0 = blockIdx.y * 128;

    // async-copy work split: 1024 16B-chunks per tile; thread owns
    // rows rB, rB+32, rB+64, rB+96 at chunk column cq (8 ushorts each)
    const int rB = tid >> 3;         // 0..31
    const int cq = tid & 7;          // 0..7

    v8f acc[2][4];
    #pragma unroll
    for (int mi = 0; mi < 2; ++mi)
        #pragma unroll
        for (int ni = 0; ni < 4; ++ni)
            acc[mi][ni] = (v8f)(0.0f);

    // ---- prologue: async-prefetch K-tile 0 into buffer 0
    #pragma unroll
    for (int i = 0; i < 4; ++i) {
        int row = rB + 32 * i;
        async_copy_b128((unsigned)(size_t)&lA[0][row * LDSROW + cq * 8],
                        Xbf + (size_t)(r0 + row) * DIN + cq * 8);
        async_copy_b128((unsigned)(size_t)&lB[0][row * LDSROW + cq * 8],
                        Wbf + (size_t)(n0 + row) * DIN + cq * 8);
    }

    int buf = 0;
    for (int k0 = 0; k0 < DIN; k0 += 64) {
        wait_async0();               // this wave's 8 chunks for tile k0 landed
        __syncthreads();             // everyone's chunks landed

        // ---- load all fragments for both K-substeps (24x ds_load_b128)
        FragU a[2][2];               // [substep][mi]
        FragU bf[2][4];              // [substep][ni]
        #pragma unroll
        for (int s = 0; s < 2; ++s) {
            #pragma unroll
            for (int mi = 0; mi < 2; ++mi) {
                int base = (wy * 32 + mi * 16 + l15) * LDSROW + s * 32;
                a[s][mi].q[0] = *(const uint4*)&lA[buf][base + g * 8];
                a[s][mi].q[1] = *(const uint4*)&lA[buf][base + 16 + g * 8];
            }
            #pragma unroll
            for (int ni = 0; ni < 4; ++ni) {
                int base = (wx * 64 + ni * 16 + l15) * LDSROW + s * 32;
                bf[s][ni].q[0] = *(const uint4*)&lB[buf][base + g * 8];
                bf[s][ni].q[1] = *(const uint4*)&lB[buf][base + 16 + g * 8];
            }
        }
        wait_dscnt0();               // fragments retired into VGPRs
        __syncthreads();             // all waves done reading this buffer

        // ---- async-prefetch next K-tile into the vacated buffer
        if (k0 + 64 < DIN) {
            #pragma unroll
            for (int i = 0; i < 4; ++i) {
                int row = rB + 32 * i;
                async_copy_b128(
                    (unsigned)(size_t)&lA[buf ^ 1][row * LDSROW + cq * 8],
                    Xbf + (size_t)(r0 + row) * DIN + (k0 + 64) + cq * 8);
                async_copy_b128(
                    (unsigned)(size_t)&lB[buf ^ 1][row * LDSROW + cq * 8],
                    Wbf + (size_t)(n0 + row) * DIN + (k0 + 64) + cq * 8);
            }
        }

        // ---- 16 WMMAs, overlapping the in-flight async prefetch
        #pragma unroll
        for (int s = 0; s < 2; ++s)
            #pragma unroll
            for (int mi = 0; mi < 2; ++mi)
                #pragma unroll
                for (int ni = 0; ni < 4; ++ni)
                    acc[mi][ni] = __builtin_amdgcn_wmma_f32_16x16x32_bf16(
                        false, a[s][mi].v, false, bf[s][ni].v,
                        (short)0, acc[mi][ni], false, false);

        buf ^= 1;
    }

    // ---- epilogue: + bias, store f32 psp
    float bcol[4];
    #pragma unroll
    for (int ni = 0; ni < 4; ++ni)
        bcol[ni] = bias[n0 + wx * 64 + ni * 16 + l15];

    #pragma unroll
    for (int mi = 0; mi < 2; ++mi) {
        #pragma unroll
        for (int ni = 0; ni < 4; ++ni) {
            int ncol = n0 + wx * 64 + ni * 16 + l15;
            #pragma unroll
            for (int r = 0; r < 8; ++r) {
                int m = r0 + wy * 32 + mi * 16 + g * 8 + r;
                psp[(size_t)m * NN + ncol] = acc[mi][ni][r] + bcol[ni];
            }
        }
    }
}

// ---------------------------------------------------------------------------
// Kernel 4: per-(t,n) batch statistics over B=128 (biased var, like jnp.var)
// ---------------------------------------------------------------------------
__global__ void snn_stats(const float* __restrict__ psp,
                          float* __restrict__ mean,
                          float* __restrict__ rstd) {
    int idx = blockIdx.x * 256 + threadIdx.x;     // idx = t*N + n
    if (idx >= TT * NN) return;
    int t = idx >> 11;
    int n = idx & (NN - 1);
    const float* p = psp + (size_t)t * BB * NN + n;
    float s = 0.0f, ss = 0.0f;
    #pragma unroll 4
    for (int b = 0; b < BB; ++b) {
        float v = p[(size_t)b * NN];              // coalesced across n
        s += v;
        ss += v * v;
    }
    float m   = s * (1.0f / BB);
    float var = ss * (1.0f / BB) - m * m;
    mean[idx] = m;
    rstd[idx] = rsqrtf(var + BN_EPS);
}

// ---------------------------------------------------------------------------
// Kernel 5: membrane scan over t per (b,n); writes spikes + final (v, reset)
// d_out layout: spikes[B][N][T] | final_v[B][N] | final_reset[B][N]
// ---------------------------------------------------------------------------
__global__ void snn_scan(const float* __restrict__ psp,
                         const float* __restrict__ mean,
                         const float* __restrict__ rstd,
                         const float* __restrict__ gamma,
                         const float* __restrict__ decay_v,
                         const float* __restrict__ reset_decay,
                         const float* __restrict__ reset_v,
                         float* __restrict__ out) {
    int idx = blockIdx.x * 256 + threadIdx.x;     // idx = b*N + n
    if (idx >= BB * NN) return;
    int n = idx & (NN - 1);
    int b = idx >> 11;

    const float dv = decay_v[n];
    const float rd = reset_decay[n];
    const float rv = reset_v[n];

    float v = 0.0f, rst = 0.0f;
    float* sp = out + (size_t)idx * TT;

    for (int t = 0; t < TT; ++t) {
        float p  = psp[((size_t)t * BB + b) * NN + n];   // coalesced across n
        int   tn = t * NN + n;
        float bn = gamma[tn] * (p - mean[tn]) * rstd[tn];
        v = v * dv + bn - rst;
        float s = (v > 1.0f) ? 1.0f : 0.0f;
        rst = rst * rd + s * rv;
        sp[t] = s;
    }
    const size_t base = (size_t)BB * NN * TT;
    out[base + idx]            = v;
    out[base + BB * NN + idx]  = rst;
}

// ---------------------------------------------------------------------------
extern "C" void kernel_launch(void* const* d_in, const int* in_sizes, int n_in,
                              void* d_out, int out_size, void* d_ws, size_t ws_size,
                              hipStream_t stream) {
    const float* in          = (const float*)d_in[0];  // [B, DIN, T]
    const float* W           = (const float*)d_in[1];  // [N, DIN]
    const float* bias        = (const float*)d_in[2];  // [N]
    const float* gamma       = (const float*)d_in[3];  // [T, N]
    const float* decay_v     = (const float*)d_in[4];  // [N]
    const float* reset_decay = (const float*)d_in[5];  // [N]
    const float* reset_v     = (const float*)d_in[6];  // [N]
    float* out = (float*)d_out;

    // workspace carve-up
    char* ws = (char*)d_ws;
    unsigned short* Xbf = (unsigned short*)ws;                       // 6400*2048*2  = 26,214,400 B
    unsigned short* Wbf = (unsigned short*)(ws + (size_t)26214400);  // 2048*2048*2  =  8,388,608 B
    float* psp  = (float*)(ws + (size_t)26214400 + 8388608);         // 6400*2048*4  = 52,428,800 B
    float* mean = (float*)(ws + (size_t)26214400 + 8388608 + 52428800);
    float* rstd = mean + TT * NN;

    // 1) pack W to bf16
    snn_pack_w<<<(NN * DIN + 255) / 256, 256, 0, stream>>>(W, Wbf, NN * DIN);

    // 2) transpose+pack X to time-major bf16
    dim3 gx(DIN / 32, (TT + 31) / 32, BB);
    dim3 bx(32, 8);
    snn_pack_x<<<gx, bx, 0, stream>>>(in, Xbf);

    // 3) fused GEMM: [6400 x 2048] = [6400 x 2048] x [2048 x 2048]^T
    dim3 gg(NN / 128, MROWS / 128);   // (16, 50)
    snn_gemm_psp<<<gg, 256, 0, stream>>>(Xbf, Wbf, bias, psp);

    // 4) batch-norm statistics per (t, n)
    snn_stats<<<(TT * NN + 255) / 256, 256, 0, stream>>>(psp, mean, rstd);

    // 5) sequential membrane scan, spike output + final state
    snn_scan<<<(BB * NN + 255) / 256, 256, 0, stream>>>(
        psp, mean, rstd, gamma, decay_v, reset_decay, reset_v, out);
}
